// GatedGCN_72018011619926
// MI455X (gfx1250) — compile-verified, matrix operand order checked by reading, not run on hardware
//
#include <hip/hip_runtime.h>
#include <hip/hip_bf16.h>
#include <stdint.h>
#include <stddef.h>

// ---------------- problem constants ----------------
constexpr int NNODE = 20000;
constexpr int NEDGE = 320000;
constexpr int CCH   = 256;   // hidden channels
constexpr int C3    = 768;   // 3*C
constexpr int IND   = 128;   // input features
constexpr int OUTD  = 128;   // output features
constexpr int LLAY  = 3;     // gated blocks
constexpr int NGR   = 16;    // graphs
constexpr float GN_EPS = 1e-5f;

// ---------------- WMMA types ----------------
typedef __bf16        bf16x16 __attribute__((ext_vector_type(16)));
typedef float         f32x8   __attribute__((ext_vector_type(8)));
typedef unsigned int  u32x4   __attribute__((ext_vector_type(4)));

union FragU { u32x4 u[2]; bf16x16 v; };

// ---------------- helpers ----------------
__device__ __forceinline__ unsigned short f2bf(float f) {
    unsigned int u = __float_as_uint(f);
    unsigned int r = (u + 0x7FFFu + ((u >> 16) & 1u)) >> 16;   // RNE
    return (unsigned short)r;
}

// ================= bf16 WMMA GEMM =================
// C[M x Nc] = A[M x K](bf16,row-major) * Bt[Nc x K](bf16,row-major, Bt = B^T) + bias
// block = 256 threads (8 waves); wave -> 32x64 strip (2 M-tiles share each B fragment);
// grid = (ceil(M/256), Nc/64). K multiple of 32; M multiple of 16.
__global__ __launch_bounds__(256) void gemm_bf16_wmma(
    const unsigned short* __restrict__ A, const unsigned short* __restrict__ Bt,
    const float* __restrict__ bias, float* __restrict__ Cout,
    int M, int K, int Ncols)
{
    const int wave  = threadIdx.x >> 5;
    const int lane  = threadIdx.x & 31;
    const int half  = lane >> 4;
    const int l16   = lane & 15;
    const int mBase = blockIdx.x * 256 + wave * 32;
    const int nBase = blockIdx.y * 64;
    if (mBase >= M) return;                 // uniform per wave: EXEC stays all-ones
    const bool two = (mBase + 16) < M;      // second M-tile valid?

    f32x8 acc0[4] = {};
    f32x8 acc1[4] = {};
    const unsigned short* Arow0 = A + (size_t)(mBase + l16) * K;
    const unsigned short* Arow1 = two ? (A + (size_t)(mBase + 16 + l16) * K) : Arow0;

    for (int k0 = 0; k0 < K; k0 += 32) {
        FragU a0, a1;
        a0.u[0] = *(const u32x4*)(Arow0 + k0 + half * 8);        // K pairs 0..7 / 8..15
        a0.u[1] = *(const u32x4*)(Arow0 + k0 + 16 + half * 8);   // K pairs 16..23 / 24..31
        a1.u[0] = *(const u32x4*)(Arow1 + k0 + half * 8);
        a1.u[1] = *(const u32x4*)(Arow1 + k0 + 16 + half * 8);
#pragma unroll
        for (int t = 0; t < 4; ++t) {
            const unsigned short* Br = Bt + (size_t)(nBase + t * 16 + l16) * K;
            FragU b;
            b.u[0] = *(const u32x4*)(Br + k0 + half * 8);
            b.u[1] = *(const u32x4*)(Br + k0 + 16 + half * 8);
            acc0[t] = __builtin_amdgcn_wmma_f32_16x16x32_bf16(
                          false, a0.v, false, b.v, (short)0, acc0[t], false, false);
            acc1[t] = __builtin_amdgcn_wmma_f32_16x16x32_bf16(
                          false, a1.v, false, b.v, (short)0, acc1[t], false, false);
        }
    }
#pragma unroll
    for (int t = 0; t < 4; ++t) {
        const int col = nBase + t * 16 + l16;
        const float bv = bias ? bias[col] : 0.0f;
#pragma unroll
        for (int r = 0; r < 8; ++r) {
            const int row0 = mBase + half * 8 + r;               // C/D layout: VGPR r -> M=r / r+8
            Cout[(size_t)row0 * Ncols + col] = acc0[t][r] + bv;
            if (two)
                Cout[(size_t)(row0 + 16) * Ncols + col] = acc1[t][r] + bv;
        }
    }
}

// ================= conversion kernels =================
__global__ void conv_f32_bf16(const float* __restrict__ in, unsigned short* __restrict__ outp, size_t n) {
    for (size_t i = blockIdx.x * (size_t)blockDim.x + threadIdx.x; i < n;
         i += (size_t)gridDim.x * blockDim.x)
        outp[i] = f2bf(in[i]);
}

// out[c*R + r] = bf16(in[r*Cc + c])   (Bt = W^T for m = g @ W)
__global__ void transpose_conv_bf16(const float* __restrict__ in, unsigned short* __restrict__ outp,
                                    int R, int Cc) {
    int idx = blockIdx.x * blockDim.x + threadIdx.x;
    if (idx >= R * Cc) return;
    int r = idx / Cc, c = idx % Cc;
    outp[(size_t)c * R + r] = f2bf(in[(size_t)r * Cc + c]);
}

__global__ void zero_f32(float* __restrict__ p, size_t n) {
    for (size_t i = blockIdx.x * (size_t)blockDim.x + threadIdx.x; i < n;
         i += (size_t)gridDim.x * blockDim.x) p[i] = 0.0f;
}

// ================= degree / GCN =================
__global__ void deg_init(float* __restrict__ d, int n) {
    int i = blockIdx.x * blockDim.x + threadIdx.x;
    if (i < n) d[i] = 1.0f;                                     // self-loop
}
__global__ void deg_count(const int* __restrict__ col, float* __restrict__ d, int e) {
    int i = blockIdx.x * blockDim.x + threadIdx.x;
    if (i < e) atomicAdd(&d[col[i]], 1.0f);
}
__global__ void deg_rsqrt(float* __restrict__ d, int n) {
    int i = blockIdx.x * blockDim.x + threadIdx.x;
    if (i < n) d[i] = rsqrtf(d[i]);
}
// out[n,c] = bias[c] + hlin[n,c]*dinv[n]^2     (self-loop contribution + bias)
__global__ __launch_bounds__(CCH) void gcn_selfloop(const float* __restrict__ hlin,
                                                    const float* __restrict__ dinv,
                                                    const float* __restrict__ bias,
                                                    float* __restrict__ outx) {
    int n = blockIdx.x, c = threadIdx.x;
    float di = dinv[n];
    outx[(size_t)n * CCH + c] = bias[c] + hlin[(size_t)n * CCH + c] * di * di;
}
// out[col[e],c] += hlin[row[e],c] * dinv[row]*dinv[col]
__global__ __launch_bounds__(CCH) void gcn_edge_scatter(const float* __restrict__ hlin,
                                                        const int* __restrict__ row,
                                                        const int* __restrict__ col,
                                                        const float* __restrict__ dinv,
                                                        float* __restrict__ outx) {
    int e = blockIdx.x, c = threadIdx.x;
    int s = row[e], d = col[e];
    float w = dinv[s] * dinv[d];
    atomicAdd(&outx[(size_t)d * CCH + c], hlin[(size_t)s * CCH + c] * w);
}
// a[dst[e],c] += m[src[e],c]
__global__ __launch_bounds__(CCH) void edge_scatter_add(const float* __restrict__ m,
                                                        const int* __restrict__ src,
                                                        const int* __restrict__ dst,
                                                        float* __restrict__ a) {
    int e = blockIdx.x, c = threadIdx.x;
    atomicAdd(&a[(size_t)dst[e] * CCH + c], m[(size_t)src[e] * CCH + c]);
}

// ================= GraphNorm (batch is sorted) =================
constexpr int GN_NPB = 64;  // nodes per block in reduction passes
__global__ __launch_bounds__(CCH) void gn_sum(const float* __restrict__ x,
                                              const int* __restrict__ batch,
                                              float* __restrict__ sum, float* __restrict__ cnt) {
    int c = threadIdx.x;
    int n0 = blockIdx.x * GN_NPB;
    int n1 = (n0 + GN_NPB < NNODE) ? n0 + GN_NPB : NNODE;
    float acc = 0.f, cc = 0.f; int g = -1;
    for (int i = n0; i < n1; ++i) {
        int bg = batch[i];
        if (bg != g) {
            if (g >= 0) { atomicAdd(&sum[g * CCH + c], acc); if (c == 0) atomicAdd(&cnt[g], cc); }
            acc = 0.f; cc = 0.f; g = bg;
        }
        acc += x[(size_t)i * CCH + c];
        cc  += 1.f;
    }
    if (g >= 0) { atomicAdd(&sum[g * CCH + c], acc); if (c == 0) atomicAdd(&cnt[g], cc); }
}
__global__ void gn_mean(const float* __restrict__ sum, const float* __restrict__ cnt,
                        float* __restrict__ mean) {
    int i = blockIdx.x * blockDim.x + threadIdx.x;
    if (i < NGR * CCH) {
        float c = cnt[i / CCH]; c = (c < 1.f) ? 1.f : c;
        mean[i] = sum[i] / c;
    }
}
__global__ __launch_bounds__(CCH) void gn_center_var(const float* __restrict__ x,
                                                     const int* __restrict__ batch,
                                                     const float* __restrict__ mean,
                                                     const float* __restrict__ ms,
                                                     float* __restrict__ cent,
                                                     float* __restrict__ vsum) {
    int c = threadIdx.x;
    int n0 = blockIdx.x * GN_NPB;
    int n1 = (n0 + GN_NPB < NNODE) ? n0 + GN_NPB : NNODE;
    float msc = ms[c];
    float acc = 0.f; int g = -1; float mc = 0.f;
    for (int i = n0; i < n1; ++i) {
        int bg = batch[i];
        if (bg != g) {
            if (g >= 0) atomicAdd(&vsum[g * CCH + c], acc);
            acc = 0.f; g = bg; mc = mean[g * CCH + c];
        }
        float v = x[(size_t)i * CCH + c] - mc * msc;
        cent[(size_t)i * CCH + c] = v;
        acc += v * v;
    }
    if (g >= 0) atomicAdd(&vsum[g * CCH + c], acc);
}
__global__ void gn_invstd(float* __restrict__ vsum, const float* __restrict__ cnt) {
    int i = blockIdx.x * blockDim.x + threadIdx.x;
    if (i < NGR * CCH) {
        float c = cnt[i / CCH]; c = (c < 1.f) ? 1.f : c;
        vsum[i] = rsqrtf(vsum[i] / c + GN_EPS);
    }
}
// y = (res?res:0) + w*cent*invstd + b ; optionally also emit bf16(y)
__global__ void gn_apply(const float* __restrict__ cent, const int* __restrict__ batch,
                         const float* __restrict__ invstd, const float* __restrict__ w,
                         const float* __restrict__ b, const float* __restrict__ res,
                         float* __restrict__ outx, unsigned short* __restrict__ out_bf) {
    size_t total = (size_t)NNODE * CCH;
    for (size_t i = blockIdx.x * (size_t)blockDim.x + threadIdx.x; i < total;
         i += (size_t)gridDim.x * blockDim.x) {
        int n = (int)(i / CCH), c = (int)(i % CCH);
        int g = batch[n];
        float v = w[c] * cent[i] * invstd[g * CCH + c] + b[c];
        float o = (res ? res[i] : 0.0f) + v;
        outx[i] = o;
        if (out_bf) out_bf[i] = f2bf(o);
    }
}

// ================= GRU / GELU =================
// h = (1-z)*n + z*h ; also emits bf16(h) for the next GEMM (fused convert)
__global__ void gru_kernel(const float* __restrict__ gi, const float* __restrict__ gh,
                           float* __restrict__ h, unsigned short* __restrict__ out_bf) {
    size_t total = (size_t)NNODE * CCH;
    for (size_t i = blockIdx.x * (size_t)blockDim.x + threadIdx.x; i < total;
         i += (size_t)gridDim.x * blockDim.x) {
        size_t n = i / CCH; int c = (int)(i % CCH);
        size_t base = n * C3 + c;
        float ir = gi[base], iz = gi[base + CCH], inn = gi[base + 2 * CCH];
        float hr = gh[base], hz = gh[base + CCH], hnn = gh[base + 2 * CCH];
        float r = 1.0f / (1.0f + __expf(-(ir + hr)));
        float z = 1.0f / (1.0f + __expf(-(iz + hz)));
        float nv = tanhf(inn + r * hnn);
        float o = (1.0f - z) * nv + z * h[i];
        h[i] = o;
        out_bf[i] = f2bf(o);
    }
}
__global__ void gelu_kernel(float* __restrict__ x, size_t n) {
    for (size_t i = blockIdx.x * (size_t)blockDim.x + threadIdx.x; i < n;
         i += (size_t)gridDim.x * blockDim.x) {
        float v = x[i];
        x[i] = 0.5f * v * (1.0f + erff(v * 0.70710678118654752f));
    }
}

// ================= host-side helpers =================
static void launch_zero(float* p, size_t n, hipStream_t s) {
    int blocks = (int)((n + 255) / 256); if (blocks > 2048) blocks = 2048;
    zero_f32<<<blocks, 256, 0, s>>>(p, n);
}
static void launch_conv(const float* in, unsigned short* outp, size_t n, hipStream_t s) {
    int blocks = (int)((n + 255) / 256); if (blocks > 4096) blocks = 4096;
    conv_f32_bf16<<<blocks, 256, 0, s>>>(in, outp, n);
}
static void launch_gemm(const unsigned short* A, const unsigned short* Bt, const float* bias,
                        float* C, int M, int K, int Nc, hipStream_t s) {
    dim3 grid((M + 255) / 256, Nc / 64);
    gemm_bf16_wmma<<<grid, 256, 0, s>>>(A, Bt, bias, C, M, K, Nc);
}
static void run_graphnorm(const float* x, const int* batch, const float* w, const float* b,
                          const float* ms, const float* res, float* outx, unsigned short* out_bf,
                          float* cent, float* sum, float* mean, float* vsum, float* cnt,
                          hipStream_t s) {
    launch_zero(sum, (size_t)NGR * CCH, s);
    launch_zero(vsum, (size_t)NGR * CCH, s);
    launch_zero(cnt, NGR, s);
    int rb = (NNODE + GN_NPB - 1) / GN_NPB;
    gn_sum<<<rb, CCH, 0, s>>>(x, batch, sum, cnt);
    gn_mean<<<(NGR * CCH + 255) / 256, 256, 0, s>>>(sum, cnt, mean);
    gn_center_var<<<rb, CCH, 0, s>>>(x, batch, mean, ms, cent, vsum);
    gn_invstd<<<(NGR * CCH + 255) / 256, 256, 0, s>>>(vsum, cnt);
    gn_apply<<<2048, 256, 0, s>>>(cent, batch, vsum, w, b, res, outx, out_bf);
}

// ================= entry point =================
extern "C" void kernel_launch(void* const* d_in, const int* in_sizes, int n_in,
                              void* d_out, int out_size, void* d_ws, size_t ws_size,
                              hipStream_t stream) {
    const float* x       = (const float*)d_in[0];
    const int*   ei      = (const int*)d_in[1];      // [2, E]
    const int*   batch   = (const int*)d_in[2];
    const float* gcn_w   = (const float*)d_in[3];    // [C, IN]
    const float* gcn_b   = (const float*)d_in[4];
    const float* gn0_w   = (const float*)d_in[5];
    const float* gn0_b   = (const float*)d_in[6];
    const float* gn0_ms  = (const float*)d_in[7];
    const float* ggc_w   = (const float*)d_in[8];    // [L,2,C,C]
    const float* gru_wih = (const float*)d_in[9];    // [L,3C,C]
    const float* gru_whh = (const float*)d_in[10];   // [L,3C,C]
    const float* gru_bih = (const float*)d_in[11];   // [L,3C]
    const float* gru_bhh = (const float*)d_in[12];   // [L,3C]
    const float* gn_w    = (const float*)d_in[13];   // [L,C]
    const float* gn_b    = (const float*)d_in[14];
    const float* gn_ms   = (const float*)d_in[15];
    const float* lin_w   = (const float*)d_in[16];   // [OUT, C]
    const float* lin_b   = (const float*)d_in[17];
    float* outp = (float*)d_out;

    const int* e_src = ei;            // edge_index[0]
    const int* e_dst = ei + NEDGE;    // edge_index[1]

    // ---- carve workspace ----
    char* p = (char*)d_ws;
    auto carve = [&](size_t bytes) -> char* {
        char* r = p; p += (bytes + 255) & ~(size_t)255; return r;
    };
    float* hb    = (float*)carve((size_t)NNODE * CCH * 4);      // h
    float* gb    = (float*)carve((size_t)NNODE * CCH * 4);      // g
    float* tmp1  = (float*)carve((size_t)NNODE * CCH * 4);      // m / cent
    float* aggb  = (float*)carve((size_t)NNODE * CCH * 4);      // a
    float* gib   = (float*)carve((size_t)NNODE * C3 * 4);       // gi
    float* ghb   = (float*)carve((size_t)NNODE * C3 * 4);       // gh
    unsigned short* gbf = (unsigned short*)carve((size_t)NNODE * CCH * 2);  // bf16(g/h)
    unsigned short* abf = (unsigned short*)carve((size_t)NNODE * CCH * 2);  // bf16(a/x)
    float* dinv  = (float*)carve((size_t)NNODE * 4);
    float* sumb  = (float*)carve((size_t)NGR * CCH * 4);
    float* meanb = (float*)carve((size_t)NGR * CCH * 4);
    float* vsumb = (float*)carve((size_t)NGR * CCH * 4);
    float* cntb  = (float*)carve((size_t)NGR * 4);
    unsigned short* gcnw_bf = (unsigned short*)carve((size_t)CCH * IND * 2);
    unsigned short* ggcT_bf = (unsigned short*)carve((size_t)LLAY * 2 * CCH * CCH * 2);
    unsigned short* wih_bf  = (unsigned short*)carve((size_t)LLAY * C3 * CCH * 2);
    unsigned short* whh_bf  = (unsigned short*)carve((size_t)LLAY * C3 * CCH * 2);
    unsigned short* lin_bf  = (unsigned short*)carve((size_t)OUTD * CCH * 2);

    // ---- convert weights to bf16 (Bt layouts) ----
    launch_conv(gcn_w, gcnw_bf, (size_t)CCH * IND, stream);        // Bt = gcn_w (h = x @ gcn_w^T)
    for (int li = 0; li < LLAY * 2; ++li)                          // Bt = ggc_w^T (m = g @ W)
        transpose_conv_bf16<<<(CCH * CCH + 255) / 256, 256, 0, stream>>>(
            ggc_w + (size_t)li * CCH * CCH, ggcT_bf + (size_t)li * CCH * CCH, CCH, CCH);
    launch_conv(gru_wih, wih_bf, (size_t)LLAY * C3 * CCH, stream); // Bt = wih
    launch_conv(gru_whh, whh_bf, (size_t)LLAY * C3 * CCH, stream); // Bt = whh
    launch_conv(lin_w, lin_bf, (size_t)OUTD * CCH, stream);        // Bt = lin_w

    // ---- degrees: deg = 1 (self loop) + indegree; dinv = rsqrt(deg) ----
    deg_init<<<(NNODE + 255) / 256, 256, 0, stream>>>(dinv, NNODE);
    deg_count<<<(NEDGE + 255) / 256, 256, 0, stream>>>(e_dst, dinv, NEDGE);
    deg_rsqrt<<<(NNODE + 255) / 256, 256, 0, stream>>>(dinv, NNODE);

    // ---- GCNConv: tmp1 = x @ gcn_w^T ; gb = b + selfloop + edge scatter ----
    launch_conv(x, abf, (size_t)NNODE * IND, stream);
    launch_gemm(abf, gcnw_bf, nullptr, tmp1, NNODE, IND, CCH, stream);
    gcn_selfloop<<<NNODE, CCH, 0, stream>>>(tmp1, dinv, gcn_b, gb);
    gcn_edge_scatter<<<NEDGE, CCH, 0, stream>>>(tmp1, e_src, e_dst, dinv, gb);

    // ---- GraphNorm0 -> hb (+ bf16 into gbf, consumed by layer 0) ----
    run_graphnorm(gb, batch, gn0_w, gn0_b, gn0_ms, nullptr, hb, gbf,
                  tmp1, sumb, meanb, vsumb, cntb, stream);

    // ---- gated blocks ----
    for (int l = 0; l < LLAY; ++l) {
        hipMemcpyAsync(gb, hb, (size_t)NNODE * CCH * 4, hipMemcpyDeviceToDevice, stream);
        for (int i = 0; i < 2; ++i) {
            // gbf currently holds bf16(g): from gn_apply (i==0) or gru_kernel (i==1)
            launch_gemm(gbf, ggcT_bf + (size_t)(l * 2 + i) * CCH * CCH,
                        nullptr, tmp1, NNODE, CCH, CCH, stream);               // m = g @ W
            launch_zero(aggb, (size_t)NNODE * CCH, stream);
            edge_scatter_add<<<NEDGE, CCH, 0, stream>>>(tmp1, e_src, e_dst, aggb);
            launch_conv(aggb, abf, (size_t)NNODE * CCH, stream);               // bf16(a)
            launch_gemm(abf, wih_bf + (size_t)l * C3 * CCH,
                        gru_bih + (size_t)l * C3, gib, NNODE, CCH, C3, stream);// gi = a@Wih^T + bih
            launch_gemm(gbf, whh_bf + (size_t)l * C3 * CCH,
                        gru_bhh + (size_t)l * C3, ghb, NNODE, CCH, C3, stream);// gh = g@Whh^T + bhh
            gru_kernel<<<2048, 256, 0, stream>>>(gib, ghb, gb, gbf);           // g = GRU(a, g), + bf16
        }
        gelu_kernel<<<2048, 256, 0, stream>>>(gb, (size_t)NNODE * CCH);
        run_graphnorm(gb, batch, gn_w + (size_t)l * CCH, gn_b + (size_t)l * CCH,
                      gn_ms + (size_t)l * CCH, /*res=*/hb, /*out=*/hb, /*bf16=*/gbf,
                      tmp1, sumb, meanb, vsumb, cntb, stream);                 // h += GN(gelu(g))
    }

    // ---- final linear -> d_out (gbf holds bf16(h) from last gn_apply) ----
    launch_gemm(gbf, lin_bf, lin_b, outp, NNODE, CCH, OUTD, stream);
}
